// cruw_61744449847352
// MI455X (gfx1250) — matrix operand order repaired and checked by reading, not compiled, not run on hardware
//
// MI455X (gfx1250, wave32) Mamba + physical-pos-embed block.
// GEMMs: v_wmma_f32_16x16x32_bf16 (bf16 in, f32 accum), TN layout, 64x16 per wave.
// Scan: chunked linear-recurrence (3-pass) for 16x block parallelism; states in
// VGPRs, B/C vectors broadcast from LDS. Pos-embed normalization cancels
// analytically, so batch_params / has_velocity are provably unused.
#include <hip/hip_runtime.h>
#include <hip/hip_bf16.h>

#define BD 8
#define LD 1024
#define DD 768
#define ND 16
#define RD 48
#define KC 4
#define MROWS (BD * LD)   // 8192 token rows
#define CH 16             // scan chunks
#define CL (LD / CH)      // 64 steps per chunk

typedef __attribute__((ext_vector_type(16))) __bf16 v16bf;
typedef __attribute__((ext_vector_type(8)))  float  v8f;

__device__ __forceinline__ unsigned short f2bf(float f) {
  unsigned int u = __float_as_uint(f);
  unsigned int r = u + 0x7fffu + ((u >> 16) & 1u);   // round-to-nearest-even
  return (unsigned short)(r >> 16);
}

// Build a 16-element bf16 fragment from two aligned 16-byte chunks.
__device__ __forceinline__ v16bf make_frag(const unsigned short* p0,
                                           const unsigned short* p1) {
  union U { uint4 q; unsigned short s[8]; } a, b;
  a.q = *(const uint4*)p0;
  b.q = *(const uint4*)p1;
  v16bf f;
#pragma unroll
  for (int i = 0; i < 8; ++i) {
    f[i]     = __builtin_bit_cast(__bf16, a.s[i]);
    f[i + 8] = __builtin_bit_cast(__bf16, b.s[i]);
  }
  return f;
}

// ---------------------------------------------------------------------------
// f32 -> bf16 with optional column zero-padding (for dt_proj K=48 -> 64).
// ---------------------------------------------------------------------------
__global__ void cvt_pad_kernel(const float* __restrict__ src,
                               unsigned short* __restrict__ dst,
                               int rows, int src_ld, int src_cols, int dst_cols) {
  size_t idx = (size_t)blockIdx.x * blockDim.x + threadIdx.x;
  size_t tot = (size_t)rows * dst_cols;
  if (idx >= tot) return;
  int c = (int)(idx % dst_cols);
  int r = (int)(idx / dst_cols);
  dst[idx] = (c < src_cols) ? f2bf(src[(size_t)r * src_ld + c]) : (unsigned short)0;
}

// ---------------------------------------------------------------------------
// Pos-embed + residual + RMSNorm.  One block per token row (b*L + l).
// Normalized coords reduce analytically to (h/15, w/15, d/3).
// ---------------------------------------------------------------------------
__global__ __launch_bounds__(256) void posnorm_kernel(
    const float* __restrict__ x, const float* __restrict__ norm_w,
    float* __restrict__ resid, unsigned short* __restrict__ hid_bf) {
  __shared__ float red[256];
  int row = blockIdx.x;            // b*L + l
  int l   = row & (LD - 1);
  int tid = threadIdx.x;
  float coords[3] = { (float)(l >> 6) * (1.f / 15.f),
                      (float)((l >> 2) & 15) * (1.f / 15.f),
                      (float)(l & 3) * (1.f / 3.f) };
  float rv[3];
  float ss = 0.f;
#pragma unroll
  for (int j = 0; j < 3; ++j) {
    int d  = tid + j * 256;
    int jj = d & 255;
    int f  = jj & 127;
    float om = __expf(-(float)f * (9.210340371976184f / 128.f)); // 10000^-(f/128)
    float v  = coords[j] * om;
    float pe = (jj < 128) ? __sinf(v) : __cosf(v);
    float r  = x[(size_t)row * DD + d] + pe;
    rv[j] = r;
    ss += r * r;
  }
  red[tid] = ss;
  __syncthreads();
  for (int s = 128; s > 0; s >>= 1) {
    if (tid < s) red[tid] += red[tid + s];
    __syncthreads();
  }
  float rms = rsqrtf(red[0] * (1.f / (float)DD) + 1e-5f);
#pragma unroll
  for (int j = 0; j < 3; ++j) {
    int d = tid + j * 256;
    resid[(size_t)row * DD + d]  = rv[j];
    hid_bf[(size_t)row * DD + d] = f2bf(rv[j] * rms * norm_w[d]);
  }
}

// ---------------------------------------------------------------------------
// Generic TN GEMM on WMMA: C[M,N] = A[M,K] * Bw[N,K]^T (+bias)(+softplus)(+resid)
// Wave -> 64x16 strip (4 x 16x16 tiles, B fragment reused 4x per K-step).
// Block = 8 waves = 512 rows.  M%512==0, N%16==0, K%32==0 for our shapes.
// ---------------------------------------------------------------------------
__global__ __launch_bounds__(256) void wmma_gemm_kernel(
    const unsigned short* __restrict__ A,   // M x K bf16 row-major
    const unsigned short* __restrict__ Bw,  // N x K bf16 row-major
    const float* __restrict__ bias,         // N (or null)
    const float* __restrict__ resid,        // M x N (or null)
    float* __restrict__ C,                  // M x N f32
    int M, int N, int Kd, int act) {
  int lane = threadIdx.x & 31;
  int wave = threadIdx.x >> 5;
  int col0 = blockIdx.x * 16;
  int row0 = (blockIdx.y * 8 + wave) * 64;

  v8f acc[4] = {};

  int arow = lane & 15;
  int akb  = (lane >> 4) * 8;    // A lane K split: {0..7,16..23} / {8..15,24..31}
  int ncol = lane & 15;
  int bkb  = (lane >> 4) * 8;    // B lane K split: {0..7,16..23} / {8..15,24..31}

  const unsigned short* pA = A  + (size_t)(row0 + arow) * Kd + akb;
  const unsigned short* pB = Bw + (size_t)(col0 + ncol) * Kd + bkb;

  for (int k0 = 0; k0 < Kd; k0 += 32) {
    if (k0 + 32 < Kd) {
      __builtin_prefetch(pB + k0 + 32, 0, 1);     // -> global_prefetch_b8
      __builtin_prefetch(pA + k0 + 32, 0, 1);
    }
    v16bf bf = make_frag(pB + k0, pB + k0 + 16);
#pragma unroll
    for (int t = 0; t < 4; ++t) {
      const unsigned short* pa = pA + (size_t)t * 16 * Kd + k0;
      v16bf af = make_frag(pa, pa + 16);
      acc[t] = __builtin_amdgcn_wmma_f32_16x16x32_bf16(
          false, af, false, bf, (short)0, acc[t], false, false);
    }
  }

  int n = col0 + ncol;
  float bv = bias ? bias[n] : 0.f;
  int mbase = row0 + (lane >> 4) * 8;
#pragma unroll
  for (int t = 0; t < 4; ++t) {
#pragma unroll
    for (int r = 0; r < 8; ++r) {
      int m = mbase + t * 16 + r;
      float v = acc[t][r] + bv;
      if (act == 1) v = (v > 20.f) ? v : log1pf(__expf(v));   // softplus
      if (resid) v += resid[(size_t)m * N + n];
      C[(size_t)m * N + n] = v;
    }
  }
}

// ---------------------------------------------------------------------------
// Depthwise causal conv (K=4) + SiLU on x half of xz.
// ---------------------------------------------------------------------------
__global__ void conv_silu_kernel(const float* __restrict__ xz,
                                 const float* __restrict__ conv_w,
                                 const float* __restrict__ conv_b,
                                 float* __restrict__ xconv,
                                 unsigned short* __restrict__ xconv_bf) {
  int idx = blockIdx.x * blockDim.x + threadIdx.x;
  if (idx >= MROWS * DD) return;
  int d   = idx % DD;
  int row = idx / DD;
  int l   = row & (LD - 1);
  int rb  = row - l;               // b*L
  float s = conv_b[d];
#pragma unroll
  for (int k = 0; k < KC; ++k) {
    int ls = l + k - (KC - 1);
    if (ls >= 0) s += xz[(size_t)(rb + ls) * (2 * DD) + d] * conv_w[d * KC + k];
  }
  float sv = s / (1.f + __expf(-s));
  xconv[idx]    = sv;
  xconv_bf[idx] = f2bf(sv);
}

// ---------------------------------------------------------------------------
// Chunked selective scan.  The diagonal recurrence h_t = dA_t h_{t-1} + dBx_t
// is affine, so a 64-step chunk composes to h_out = P*h_in + S per (b,d,n).
// Pass 1 computes (P,S) per chunk (16x parallel over L); pass 2 composes the
// 16 chunk transforms sequentially (tiny); pass 3 replays each chunk from its
// entry state computing y (+D*x, *silu(z)) -> bf16.  2x exp work for 16x
// occupancy during the dominant serial phase.
// ---------------------------------------------------------------------------
__global__ __launch_bounds__(256) void scan_pass1_kernel(
    const float* __restrict__ xconv,   // M x 768
    const float* __restrict__ dt,      // M x 768
    const float* __restrict__ xdbl,    // M x 80  (B at col 48)
    const float* __restrict__ A_log,   // 768 x 16
    float* __restrict__ Pbuf,          // CH x B x 768 x 16
    float* __restrict__ Sbuf)          // CH x B x 768 x 16
{
  __shared__ float Bs[CL][16];
  int tid = threadIdx.x;
  int c   = blockIdx.x % CH;
  int g   = blockIdx.x / CH;          // b*3 + dgroup
  int b   = g / 3;
  int d   = (g % 3) * 256 + tid;
  int bL  = b * LD;
  int t0  = c * CL;

  float a[ND];
#pragma unroll
  for (int n = 0; n < ND; ++n) a[n] = -__expf(A_log[d * ND + n]);
  float P[ND], S[ND];
#pragma unroll
  for (int n = 0; n < ND; ++n) { P[n] = 1.f; S[n] = 0.f; }

  for (int i = tid; i < CL * 16; i += 256) {
    int tt = i >> 4, nn = i & 15;
    Bs[tt][nn] = xdbl[(size_t)(bL + t0 + tt) * 80 + 48 + nn];
  }
  __syncthreads();

  for (int tt = 0; tt < CL; ++tt) {
    size_t rb = (size_t)(bL + t0 + tt);
    float dtv = dt[rb * DD + d];
    float xv  = xconv[rb * DD + d];
    float dBx = dtv * xv;
#pragma unroll
    for (int n = 0; n < ND; ++n) {
      float dA = __expf(dtv * a[n]);
      P[n] *= dA;
      S[n] = dA * S[n] + dBx * Bs[tt][n];
    }
  }
  size_t base = ((size_t)c * BD * DD + (size_t)b * DD + d) * ND;
#pragma unroll
  for (int n = 0; n < ND; ++n) { Pbuf[base + n] = P[n]; Sbuf[base + n] = S[n]; }
}

__global__ void scan_pass2_kernel(const float* __restrict__ Pbuf,
                                  const float* __restrict__ Sbuf,
                                  float* __restrict__ Hin) {
  int idx = blockIdx.x * blockDim.x + threadIdx.x;   // (b*DD + d)*ND + n
  if (idx >= BD * DD * ND) return;
  const size_t stride = (size_t)BD * DD * ND;
  float h = 0.f;
  for (int c = 0; c < CH; ++c) {
    size_t o = (size_t)c * stride + idx;
    Hin[o] = h;
    h = Pbuf[o] * h + Sbuf[o];
  }
}

__global__ __launch_bounds__(256) void scan_pass3_kernel(
    const float* __restrict__ xz,      // M x 1536 (z at +768)
    const float* __restrict__ xconv,   // M x 768
    const float* __restrict__ dt,      // M x 768
    const float* __restrict__ xdbl,    // M x 80  (B at 48, C at 64)
    const float* __restrict__ A_log,   // 768 x 16
    const float* __restrict__ D_param, // 768
    const float* __restrict__ Hin,     // CH x B x 768 x 16
    unsigned short* __restrict__ ybf)  // M x 768
{
  __shared__ float Bs[CL][16];
  __shared__ float Cs[CL][16];
  int tid = threadIdx.x;
  int c   = blockIdx.x % CH;
  int g   = blockIdx.x / CH;
  int b   = g / 3;
  int d   = (g % 3) * 256 + tid;
  int bL  = b * LD;
  int t0  = c * CL;

  float a[ND];
#pragma unroll
  for (int n = 0; n < ND; ++n) a[n] = -__expf(A_log[d * ND + n]);
  float Dp = D_param[d];
  float h[ND];
  size_t hbase = ((size_t)c * BD * DD + (size_t)b * DD + d) * ND;
#pragma unroll
  for (int n = 0; n < ND; ++n) h[n] = Hin[hbase + n];

  for (int i = tid; i < CL * 16; i += 256) {
    int tt = i >> 4, nn = i & 15;
    size_t base = (size_t)(bL + t0 + tt) * 80;
    Bs[tt][nn] = xdbl[base + 48 + nn];
    Cs[tt][nn] = xdbl[base + 64 + nn];
  }
  __syncthreads();

  for (int tt = 0; tt < CL; ++tt) {
    size_t rb = (size_t)(bL + t0 + tt);
    float dtv = dt[rb * DD + d];
    float xv  = xconv[rb * DD + d];
    float dBx = dtv * xv;
    float y = 0.f;
#pragma unroll
    for (int n = 0; n < ND; ++n) {
      float dA = __expf(dtv * a[n]);
      h[n] = dA * h[n] + dBx * Bs[tt][n];
      y += h[n] * Cs[tt][n];
    }
    float zv = xz[rb * (2 * DD) + DD + d];
    float yt = (y + Dp * xv) * (zv / (1.f + __expf(-zv)));
    ybf[rb * DD + d] = f2bf(yt);
  }
}

// ---------------------------------------------------------------------------
extern "C" void kernel_launch(void* const* d_in, const int* in_sizes, int n_in,
                              void* d_out, int out_size, void* d_ws, size_t ws_size,
                              hipStream_t stream) {
  const float* x          = (const float*)d_in[0];
  // d_in[1] batch_params, d_in[2] has_velocity: cancel analytically (unused)
  const float* norm_w     = (const float*)d_in[3];
  const float* in_proj_w  = (const float*)d_in[4];
  const float* in_proj_b  = (const float*)d_in[5];
  const float* conv_w     = (const float*)d_in[6];
  const float* conv_b     = (const float*)d_in[7];
  const float* x_proj_w   = (const float*)d_in[8];
  const float* dt_proj_w  = (const float*)d_in[9];
  const float* dt_proj_b  = (const float*)d_in[10];
  const float* A_log      = (const float*)d_in[11];
  const float* D_param    = (const float*)d_in[12];
  const float* out_proj_w = (const float*)d_in[13];
  const float* out_proj_b = (const float*)d_in[14];

  char* ws = (char*)d_ws;
  size_t off = 0;
  auto alloc = [&](size_t bytes) -> void* {
    off = (off + 255) & ~(size_t)255;
    void* p = ws + off;
    off += bytes;
    return p;
  };

  const int M = MROWS;
  float*          resid    = (float*)alloc((size_t)M * DD * 4);
  unsigned short* hid_bf   = (unsigned short*)alloc((size_t)M * DD * 2);
  unsigned short* w_in_bf  = (unsigned short*)alloc((size_t)2 * DD * DD * 2);
  unsigned short* w_xp_bf  = (unsigned short*)alloc((size_t)80 * DD * 2);
  unsigned short* w_dt_bf  = (unsigned short*)alloc((size_t)DD * 64 * 2);
  unsigned short* w_out_bf = (unsigned short*)alloc((size_t)DD * DD * 2);
  float*          xz       = (float*)alloc((size_t)M * 2 * DD * 4);
  float*          xconv    = (float*)alloc((size_t)M * DD * 4);
  unsigned short* xconv_bf = (unsigned short*)alloc((size_t)M * DD * 2);
  float*          xdbl     = (float*)alloc((size_t)M * 80 * 4);
  unsigned short* dtin_bf  = (unsigned short*)alloc((size_t)M * 64 * 2);
  float*          dtb      = (float*)alloc((size_t)M * DD * 4);
  unsigned short* y_bf     = (unsigned short*)alloc((size_t)M * DD * 2);
  float*          Pbuf     = (float*)alloc((size_t)CH * BD * DD * ND * 4);
  float*          Sbuf     = (float*)alloc((size_t)CH * BD * DD * ND * 4);
  float*          Hin      = (float*)alloc((size_t)CH * BD * DD * ND * 4);

  // Weight conversions (bf16; dt_proj zero-padded K 48 -> 64)
  cvt_pad_kernel<<<(2 * DD * DD + 255) / 256, 256, 0, stream>>>(in_proj_w, w_in_bf, 2 * DD, DD, DD, DD);
  cvt_pad_kernel<<<(80 * DD + 255) / 256, 256, 0, stream>>>(x_proj_w, w_xp_bf, 80, DD, DD, DD);
  cvt_pad_kernel<<<(DD * 64 + 255) / 256, 256, 0, stream>>>(dt_proj_w, w_dt_bf, DD, RD, RD, 64);
  cvt_pad_kernel<<<(DD * DD + 255) / 256, 256, 0, stream>>>(out_proj_w, w_out_bf, DD, DD, DD, DD);

  // Pos-embed + residual + RMSNorm
  posnorm_kernel<<<M, 256, 0, stream>>>(x, norm_w, resid, hid_bf);

  // xz = hidden @ in_proj_w^T + b   (M=8192, N=1536, K=768)
  { dim3 g(2 * DD / 16, M / 512);
    wmma_gemm_kernel<<<g, 256, 0, stream>>>(hid_bf, w_in_bf, in_proj_b, nullptr, xz, M, 2 * DD, DD, 0); }

  // causal conv + SiLU
  conv_silu_kernel<<<(M * DD + 255) / 256, 256, 0, stream>>>(xz, conv_w, conv_b, xconv, xconv_bf);

  // x_dbl = x_conv @ x_proj_w^T    (N=80, K=768)
  { dim3 g(80 / 16, M / 512);
    wmma_gemm_kernel<<<g, 256, 0, stream>>>(xconv_bf, w_xp_bf, nullptr, nullptr, xdbl, M, 80, DD, 0); }

  // dt input: first 48 cols of x_dbl -> bf16, padded to 64
  cvt_pad_kernel<<<(M * 64 + 255) / 256, 256, 0, stream>>>(xdbl, dtin_bf, M, 80, RD, 64);

  // dt = softplus(dt_in @ dt_proj_w^T + b)   (N=768, K=64)
  { dim3 g(DD / 16, M / 512);
    wmma_gemm_kernel<<<g, 256, 0, stream>>>(dtin_bf, w_dt_bf, dt_proj_b, nullptr, dtb, M, DD, 64, 1); }

  // chunked selective scan (3 passes) -> y (bf16)
  scan_pass1_kernel<<<BD * 3 * CH, 256, 0, stream>>>(xconv, dtb, xdbl, A_log, Pbuf, Sbuf);
  scan_pass2_kernel<<<(BD * DD * ND + 255) / 256, 256, 0, stream>>>(Pbuf, Sbuf, Hin);
  scan_pass3_kernel<<<BD * 3 * CH, 256, 0, stream>>>(xz, xconv, dtb, xdbl, A_log, D_param, Hin, y_bf);

  // out = y @ out_proj_w^T + b + residual    (N=768, K=768)
  { dim3 g(DD / 16, M / 512);
    wmma_gemm_kernel<<<g, 256, 0, stream>>>(y_bf, w_out_bf, out_proj_b, resid, (float*)d_out, M, DD, DD, 0); }

  (void)in_sizes; (void)n_in; (void)out_size; (void)ws_size;
}
